// CausalSelfAttention_1829656068805
// MI455X (gfx1250) — matrix operand order repaired and checked by reading, not compiled, not run on hardware
//
#include <hip/hip_runtime.h>

typedef __bf16 bf16;
typedef __attribute__((ext_vector_type(16))) __bf16 v16bf;
typedef __attribute__((ext_vector_type(8)))  float  v8f;
typedef unsigned int uint32;
typedef unsigned long long uint64;

#define C_EMBD   1024
#define C3       3072
#define T_SEQ    2048
#define N_BATCH  4
#define N_HEAD   16
#define HD       64
#define M_ROWS   8192   // B*T

// ---------- bf16 helpers (native hardware convert) ----------
__device__ __forceinline__ bf16 f2bf(float f) { return (bf16)f; }

__device__ __forceinline__ uint32 pk2bf(float lo, float hi) {
  union { bf16 h[2]; uint32 u; } p;
  p.h[0] = (bf16)lo; p.h[1] = (bf16)hi;
  return p.u;
}

union FragBF { v16bf v; uint32 u[8]; };

// A-matrix fragment (16 x 32 bf16, MxK). Lane l holds row m=l&15;
// lanes 0-15: K = {0..7,16..23}, lanes 16-31: K = {8..15,24..31}.
// two 16B runs per lane, 32B apart -> 2x ds_load_b128 when 16B-aligned.
__device__ __forceinline__ v16bf load_frag_a(const bf16* base, int stride, int lane) {
  const bf16* p = base + (lane & 15) * stride + ((lane >> 4) << 3);
  FragBF f;
#pragma unroll
  for (int j = 0; j < 4; ++j) {
    f.u[j]     = *(const uint32*)(p + 2 * j);
    f.u[4 + j] = *(const uint32*)(p + 16 + 2 * j);
  }
  return f.v;
}

// B-matrix fragment (32 x 16 bf16, KxN). Lane l holds column n=l&15;
// lanes 0-15: K = {0..7,16..23}, lanes 16-31: K = {8..15,24..31} (strided).
__device__ __forceinline__ v16bf load_frag_b(const bf16* base, int stride, int lane) {
  const bf16* p = base + (lane & 15) + ((lane >> 4) << 3) * stride;
  FragBF f;
#pragma unroll
  for (int j = 0; j < 8; ++j) {
    f.v[j]     = p[j * stride];
    f.v[8 + j] = p[(16 + j) * stride];
  }
  return f.v;
}

#define WMMA_BF16(A, B, Cacc) \
  __builtin_amdgcn_wmma_f32_16x16x32_bf16(false, (A), false, (B), (short)0, (Cacc), false, false)
#define SCHED_FENCE() __builtin_amdgcn_sched_barrier(0x16)  // DS+WMMA may not cross

// ---------- async global->LDS staging (gfx1250, ASYNCcnt-tracked) ----------
__device__ __forceinline__ void async_ld_b128(uint32 lds_off, uint64 gaddr) {
  asm volatile("global_load_async_to_lds_b128 %0, %1, off"
               :: "v"(lds_off), "v"(gaddr) : "memory");
}
__device__ __forceinline__ void wait_async0() {
  asm volatile("s_wait_asynccnt 0" ::: "memory");
}
__device__ __forceinline__ uint32 lds_off_u32(const void* p) {
  return (uint32)(uintptr_t)p;   // LDS aperture: addr[31:0] is the LDS byte address
}

// ---------------------------------------------------------------------------
// GEMM tiling: block 128(M) x 128(N), 8 waves, wave tile 32x64 (8 WMMA/step),
// K-step 32, double-buffered LDS, one barrier per step.
// ---------------------------------------------------------------------------
#define SA 40    // As row stride (bf16): 80B rows, 16B-aligned, uniform 2x banks
#define SB 136   // Bs row stride (bf16): 272B rows, 16B-aligned, uniform 2x banks

struct Tile8 { float4 a[4]; float4 b[4]; };

__global__ __launch_bounds__(256) void qkv_gemm_kernel(
    const float* __restrict__ x, const float* __restrict__ W,
    const float* __restrict__ bias,
    bf16* __restrict__ qo, bf16* __restrict__ ko, bf16* __restrict__ vo) {
  __shared__ bf16 As[2][128 * SA];
  __shared__ bf16 Bs[2][32 * SB];

  const int tid  = threadIdx.x, lane = tid & 31, wave = tid >> 5;
  const int m0   = blockIdx.y * 128;
  const int n0   = blockIdx.x * 128;
  const int arow = tid >> 1, acol = (tid & 1) * 16;   // A: 2 thr/row, 16 cols
  const int bkr  = tid >> 3, bnc = (tid & 7) * 16;    // B: 8 thr/row, 16 cols
  const int wrow = (wave & 3) * 32, wcol = (wave >> 2) * 64;
  const int halb = lane >> 4, lcol = lane & 15;

  v8f acc[2][4] = {};

  auto tload = [&](int k0) {
    Tile8 r;
    const float4* xp = (const float4*)(x + (size_t)(m0 + arow) * C_EMBD + k0 + acol);
#pragma unroll
    for (int j = 0; j < 4; ++j) r.a[j] = xp[j];
    const float4* wp = (const float4*)(W + (size_t)(k0 + bkr) * C3 + n0 + bnc);
#pragma unroll
    for (int j = 0; j < 4; ++j) r.b[j] = wp[j];
    return r;
  };
  auto tstore = [&](const Tile8& r, bf16* Ab, bf16* Bb) {
    uint32* ad = (uint32*)(Ab + arow * SA + acol);
#pragma unroll
    for (int j = 0; j < 4; ++j) {
      ad[2 * j]     = pk2bf(r.a[j].x, r.a[j].y);
      ad[2 * j + 1] = pk2bf(r.a[j].z, r.a[j].w);
    }
    uint32* bd = (uint32*)(Bb + bkr * SB + bnc);
#pragma unroll
    for (int j = 0; j < 4; ++j) {
      bd[2 * j]     = pk2bf(r.b[j].x, r.b[j].y);
      bd[2 * j + 1] = pk2bf(r.b[j].z, r.b[j].w);
    }
  };
  auto compute_tile = [&](const bf16* Ac, const bf16* Bc) {
    v16bf a0 = load_frag_a(Ac + (wrow +  0) * SA, SA, lane);
    v16bf a1 = load_frag_a(Ac + (wrow + 16) * SA, SA, lane);
    v16bf b0 = load_frag_b(Bc + wcol +  0, SB, lane);
    v16bf b1 = load_frag_b(Bc + wcol + 16, SB, lane);
    v16bf b2 = load_frag_b(Bc + wcol + 32, SB, lane);
    v16bf b3 = load_frag_b(Bc + wcol + 48, SB, lane);
    SCHED_FENCE();
    acc[0][0] = WMMA_BF16(a0, b0, acc[0][0]);
    acc[0][1] = WMMA_BF16(a0, b1, acc[0][1]);
    acc[0][2] = WMMA_BF16(a0, b2, acc[0][2]);
    acc[0][3] = WMMA_BF16(a0, b3, acc[0][3]);
    acc[1][0] = WMMA_BF16(a1, b0, acc[1][0]);
    acc[1][1] = WMMA_BF16(a1, b1, acc[1][1]);
    acc[1][2] = WMMA_BF16(a1, b2, acc[1][2]);
    acc[1][3] = WMMA_BF16(a1, b3, acc[1][3]);
  };

  tstore(tload(0), As[0], Bs[0]);
  for (int s = 0; s < 31; ++s) {
    __syncthreads();
    Tile8 r = tload((s + 1) * 32);
    compute_tile(As[s & 1], Bs[s & 1]);
    tstore(r, As[(s + 1) & 1], Bs[(s + 1) & 1]);
  }
  __syncthreads();
  compute_tile(As[1], Bs[1]);

  // epilogue: bias + scatter to [B,H,T,D] bf16 (head uniform per wave)
  const int nw    = n0 + wcol;
  const int which = nw >> 10;
  const int h     = (nw & 1023) >> 6;
  bf16* outp = (which == 0) ? qo : (which == 1) ? ko : vo;
#pragma unroll
  for (int f = 0; f < 4; ++f) {
    int   d  = f * 16 + lcol;
    float bv = bias[nw + f * 16 + lcol];
#pragma unroll
    for (int r = 0; r < 2; ++r)
#pragma unroll
      for (int vv = 0; vv < 8; ++vv) {
        int row = m0 + wrow + 16 * r + vv + 8 * halb;
        int b   = row >> 11, t = row & (T_SEQ - 1);
        outp[(((size_t)(b * N_HEAD + h)) * T_SEQ + t) * HD + d] = f2bf(acc[r][f][vv] + bv);
      }
  }
}

// ---------------------------------------------------------------------------
// Kernel 2: flash attention per (b*h).  Block = 256 query rows, 8 waves,
// wave owns 32 queries x full D=64 (16 WMMA / 32-key step).
// V async-staged (b128, ASYNCcnt); K transposed, stores at loop bottom.
// ---------------------------------------------------------------------------
#define SKT 40   // Kst[d][key] stride: 80B rows, 16B-aligned
#define SVS 72   // Vs[key][d] stride: 144B rows, 16B-aligned
#define SPP 40   // Ps per-wave stride: 80B rows

__global__ __launch_bounds__(256) void attn_kernel(
    const bf16* __restrict__ q, const bf16* __restrict__ k,
    const bf16* __restrict__ v, bf16* __restrict__ y) {
  __shared__ bf16 Kst[2][64 * SKT];    // K tile transposed: [d][key]
  __shared__ bf16 Vs[2][32 * SVS];     // V tile natural:    [key][d]
  __shared__ bf16 Ps[8 * 32 * SPP];    // P staging (C-layout -> A-layout)

  const int tid  = threadIdx.x, lane = tid & 31, wave = tid >> 5;
  const int bh   = blockIdx.y;
  const int q0   = blockIdx.x * 256;
  const int qw   = q0 + wave * 32;
  const size_t base = (size_t)bh * T_SEQ * HD;
  const int halb = lane >> 4, lcol = lane & 15;
  const int lkey = tid >> 3, ldb = (tid & 7) * 8;

  v16bf qf[2][2];
#pragma unroll
  for (int r = 0; r < 2; ++r) {
    qf[r][0] = load_frag_a(q + base + (size_t)(qw + 16 * r) * HD,      HD, lane);
    qf[r][1] = load_frag_a(q + base + (size_t)(qw + 16 * r) * HD + 32, HD, lane);
  }

  v8f accY[2][4] = {};
  float m_i[2][8], l_i[2][8];
#pragma unroll
  for (int r = 0; r < 2; ++r)
#pragma unroll
    for (int i = 0; i < 8; ++i) { m_i[r][i] = -1e30f; l_i[r][i] = 0.0f; }

  const int nsteps = (blockIdx.x + 1) * 8;

  // prologue: stage tile 0
  {
    uint64 gv = (uint64)(uintptr_t)(v + base + (size_t)lkey * HD + ldb);
    async_ld_b128(lds_off_u32(Vs[0] + lkey * SVS + ldb), gv);
    union { uint4 u; bf16 hx[8]; } kt;
    kt.u = *(const uint4*)(k + base + (size_t)lkey * HD + ldb);
#pragma unroll
    for (int j = 0; j < 8; ++j) Kst[0][(ldb + j) * SKT + lkey] = kt.hx[j];
  }

  for (int s = 0; s < nsteps; ++s) {
    const int cur = s & 1, nxt = cur ^ 1;
    const int nk = s * 32;
    wait_async0();
    __syncthreads();

    const bool hn = (s + 1 < nsteps);
    union { uint4 u; bf16 hx[8]; } kt;
    kt.u = uint4{0u, 0u, 0u, 0u};
    if (hn) {
      int nk1 = nk + 32;
      uint64 gv = (uint64)(uintptr_t)(v + base + (size_t)(nk1 + lkey) * HD + ldb);
      async_ld_b128(lds_off_u32(Vs[nxt] + lkey * SVS + ldb), gv);
      kt.u = *(const uint4*)(k + base + (size_t)(nk1 + lkey) * HD + ldb);
    }

    // S B-fragments
    v16bf kb0 = load_frag_b(Kst[cur] +  0 * SKT,      SKT, lane);  // d0-31 , keys 0-15
    v16bf kb1 = load_frag_b(Kst[cur] +  0 * SKT + 16, SKT, lane);  // d0-31 , keys 16-31
    v16bf kb2 = load_frag_b(Kst[cur] + 32 * SKT,      SKT, lane);  // d32-63, keys 0-15
    v16bf kb3 = load_frag_b(Kst[cur] + 32 * SKT + 16, SKT, lane);  // d32-63, keys 16-31
    SCHED_FENCE();

    v8f Sf[2][2] = {};
#pragma unroll
    for (int r = 0; r < 2; ++r) {
      Sf[r][0] = WMMA_BF16(qf[r][0], kb0, Sf[r][0]);
      Sf[r][0] = WMMA_BF16(qf[r][1], kb2, Sf[r][0]);
      Sf[r][1] = WMMA_BF16(qf[r][0], kb1, Sf[r][1]);
      Sf[r][1] = WMMA_BF16(qf[r][1], kb3, Sf[r][1]);
    }

    // online softmax per row tile
#pragma unroll
    for (int r = 0; r < 2; ++r)
#pragma unroll
      for (int vv = 0; vv < 8; ++vv) {
        int qrow = qw + 16 * r + vv + 8 * halb;
#pragma unroll
        for (int f = 0; f < 2; ++f) {
          int key = nk + f * 16 + lcol;
          float sv = Sf[r][f][vv] * 0.125f;              // 1/sqrt(64)
          Sf[r][f][vv] = (key <= qrow) ? sv : -1e30f;    // causal mask
        }
        float mx = fmaxf(Sf[r][0][vv], Sf[r][1][vv]);
#pragma unroll
        for (int off = 1; off < 16; off <<= 1) mx = fmaxf(mx, __shfl_xor(mx, off, 32));
        float mnew = fmaxf(m_i[r][vv], mx);
        float corr = __expf(m_i[r][vv] - mnew);
        float p0   = __expf(Sf[r][0][vv] - mnew);
        float p1   = __expf(Sf[r][1][vv] - mnew);
        Sf[r][0][vv] = p0; Sf[r][1][vv] = p1;
        float rs = p0 + p1;
#pragma unroll
        for (int off = 1; off < 16; off <<= 1) rs += __shfl_xor(rs, off, 32);
        l_i[r][vv] = l_i[r][vv] * corr + rs;
        m_i[r][vv] = mnew;
#pragma unroll
        for (int f = 0; f < 4; ++f) accY[r][f][vv] *= corr;
      }

    // reshape P: C-layout -> LDS -> A-layout (wave-local, DS in-order)
    bf16* pw = Ps + wave * 32 * SPP;
#pragma unroll
    for (int r = 0; r < 2; ++r)
#pragma unroll
      for (int f = 0; f < 2; ++f)
#pragma unroll
        for (int vv = 0; vv < 8; ++vv)
          pw[(16 * r + vv + 8 * halb) * SPP + f * 16 + lcol] = f2bf(Sf[r][f][vv]);

    // V fragments late (short live range), then P fragments
    v16bf vb0 = load_frag_b(Vs[cur] +  0, SVS, lane);
    v16bf vb1 = load_frag_b(Vs[cur] + 16, SVS, lane);
    v16bf vb2 = load_frag_b(Vs[cur] + 32, SVS, lane);
    v16bf vb3 = load_frag_b(Vs[cur] + 48, SVS, lane);
    v16bf pf0 = load_frag_a(pw,            SPP, lane);
    v16bf pf1 = load_frag_a(pw + 16 * SPP, SPP, lane);
    SCHED_FENCE();
    accY[0][0] = WMMA_BF16(pf0, vb0, accY[0][0]);
    accY[0][1] = WMMA_BF16(pf0, vb1, accY[0][1]);
    accY[0][2] = WMMA_BF16(pf0, vb2, accY[0][2]);
    accY[0][3] = WMMA_BF16(pf0, vb3, accY[0][3]);
    accY[1][0] = WMMA_BF16(pf1, vb0, accY[1][0]);
    accY[1][1] = WMMA_BF16(pf1, vb1, accY[1][1]);
    accY[1][2] = WMMA_BF16(pf1, vb2, accY[1][2]);
    accY[1][3] = WMMA_BF16(pf1, vb3, accY[1][3]);

    // deferred transposed K stores for next tile (after all DS loads)
    if (hn) {
#pragma unroll
      for (int j = 0; j < 8; ++j) Kst[nxt][(ldb + j) * SKT + lkey] = kt.hx[j];
    }
  }

  // normalize + store y as [B,T,C] bf16
  const int b = bh >> 4, h = bh & 15;
#pragma unroll
  for (int r = 0; r < 2; ++r)
#pragma unroll
    for (int vv = 0; vv < 8; ++vv) {
      float rinv = 1.0f / l_i[r][vv];
      int t = qw + 16 * r + vv + 8 * halb;
      size_t rowo = ((size_t)b * T_SEQ + t) * C_EMBD + h * HD;
#pragma unroll
      for (int c = 0; c < 4; ++c)
        y[rowo + c * 16 + lcol] = f2bf(accY[r][c][vv] * rinv);
    }
}

// ---------------------------------------------------------------------------
// Kernel 3: out = y @ W_proj + b_proj   (M=8192, N=1024, K=1024, fp32 out)
// A tile async-staged (ASYNCcnt), B tile converted; double-buffered.
// ---------------------------------------------------------------------------
__global__ __launch_bounds__(256) void proj_gemm_kernel(
    const bf16* __restrict__ A, const float* __restrict__ W,
    const float* __restrict__ bias, float* __restrict__ out) {
  __shared__ bf16 As[2][128 * SA];
  __shared__ bf16 Bs[2][32 * SB];

  const int tid  = threadIdx.x, lane = tid & 31, wave = tid >> 5;
  const int m0   = blockIdx.y * 128;
  const int n0   = blockIdx.x * 128;
  const int arow = tid >> 1, acol = (tid & 1) * 16;
  const int bkr  = tid >> 3, bnc = (tid & 7) * 16;
  const int wrow = (wave & 3) * 32, wcol = (wave >> 2) * 64;
  const int halb = lane >> 4, lcol = lane & 15;

  v8f acc[2][4] = {};

  auto a_async = [&](int k0, bf16* Ab) {
    uint64 ga = (uint64)(uintptr_t)(A + (size_t)(m0 + arow) * C_EMBD + k0 + acol);
    uint32 la = lds_off_u32(Ab + arow * SA + acol);
    async_ld_b128(la, ga);
    async_ld_b128(la + 16, ga + 16);
  };
  auto b_load = [&](int k0) {
    Tile8 r;
    const float4* wp = (const float4*)(W + (size_t)(k0 + bkr) * C_EMBD + n0 + bnc);
#pragma unroll
    for (int j = 0; j < 4; ++j) r.b[j] = wp[j];
    return r;
  };
  auto b_store = [&](const Tile8& r, bf16* Bb) {
    uint32* bd = (uint32*)(Bb + bkr * SB + bnc);
#pragma unroll
    for (int j = 0; j < 4; ++j) {
      bd[2 * j]     = pk2bf(r.b[j].x, r.b[j].y);
      bd[2 * j + 1] = pk2bf(r.b[j].z, r.b[j].w);
    }
  };
  auto compute_tile = [&](const bf16* Ac, const bf16* Bc) {
    v16bf a0 = load_frag_a(Ac + (wrow +  0) * SA, SA, lane);
    v16bf a1 = load_frag_a(Ac + (wrow + 16) * SA, SA, lane);
    v16bf b0 = load_frag_b(Bc + wcol +  0, SB, lane);
    v16bf b1 = load_frag_b(Bc + wcol + 16, SB, lane);
    v16bf b2 = load_frag_b(Bc + wcol + 32, SB, lane);
    v16bf b3 = load_frag_b(Bc + wcol + 48, SB, lane);
    SCHED_FENCE();
    acc[0][0] = WMMA_BF16(a0, b0, acc[0][0]);
    acc[0][1] = WMMA_BF16(a0, b1, acc[0][1]);
    acc[0][2] = WMMA_BF16(a0, b2, acc[0][2]);
    acc[0][3] = WMMA_BF16(a0, b3, acc[0][3]);
    acc[1][0] = WMMA_BF16(a1, b0, acc[1][0]);
    acc[1][1] = WMMA_BF16(a1, b1, acc[1][1]);
    acc[1][2] = WMMA_BF16(a1, b2, acc[1][2]);
    acc[1][3] = WMMA_BF16(a1, b3, acc[1][3]);
  };

  a_async(0, As[0]);
  b_store(b_load(0), Bs[0]);
  for (int s = 0; s < 31; ++s) {
    wait_async0();
    __syncthreads();
    a_async((s + 1) * 32, As[(s + 1) & 1]);
    Tile8 r = b_load((s + 1) * 32);
    compute_tile(As[s & 1], Bs[s & 1]);
    b_store(r, Bs[(s + 1) & 1]);
  }
  wait_async0();
  __syncthreads();
  compute_tile(As[1], Bs[1]);

#pragma unroll
  for (int f = 0; f < 4; ++f) {
    int   n  = n0 + wcol + f * 16 + lcol;
    float bv = bias[n];
#pragma unroll
    for (int r = 0; r < 2; ++r)
#pragma unroll
      for (int vv = 0; vv < 8; ++vv) {
        int row = m0 + wrow + 16 * r + vv + 8 * halb;
        out[(size_t)row * C_EMBD + n] = acc[r][f][vv] + bv;
      }
  }
}

// ---------------------------------------------------------------------------
extern "C" void kernel_launch(void* const* d_in, const int* in_sizes, int n_in,
                              void* d_out, int out_size, void* d_ws, size_t ws_size,
                              hipStream_t stream) {
  (void)in_sizes; (void)n_in; (void)out_size; (void)ws_size;
  const float* x      = (const float*)d_in[0];
  const float* W_attn = (const float*)d_in[1];
  const float* b_attn = (const float*)d_in[2];
  const float* W_proj = (const float*)d_in[3];
  const float* b_proj = (const float*)d_in[4];
  float* out = (float*)d_out;

  bf16* ws = (bf16*)d_ws;
  bf16* qo = ws;
  bf16* ko = ws + 8388608u;
  bf16* vo = ws + 16777216u;
  bf16* yo = ws + 25165824u;

  qkv_gemm_kernel<<<dim3(C3 / 128, M_ROWS / 128), 256, 0, stream>>>(
      x, W_attn, b_attn, qo, ko, vo);
  attn_kernel<<<dim3(T_SEQ / 256, N_BATCH * N_HEAD), 256, 0, stream>>>(
      qo, ko, vo, yo);
  proj_gemm_kernel<<<dim3(C_EMBD / 128, M_ROWS / 128), 256, 0, stream>>>(
      yo, W_proj, b_proj, out);
}
